// FastweightMemory_1726576856739
// MI455X (gfx1250) — matrix operand order repaired and checked by reading, not compiled, not run on hardware
//
#include <hip/hip_runtime.h>

// ---------------------------------------------------------------------------
// FastweightMemory (linear-attention delta rule), MI455X / gfx1250.
// b=4, h=16, n=4096, d=64. Memory floor ~320MB -> ~14us @ 23.3TB/s.
// GEMMs (~6.5 GFLOP) run on v_wmma_f32_16x16x32_bf16 (f32 accumulate).
// Values tile staged by the Tensor Data Mover (tensor_load_to_lds,
// TENSORcnt-fenced); keys stay on the register-prefetch path because they
// need the elu featurization. Toolchain uses the 6-arg TDM builtin.
// ---------------------------------------------------------------------------

#define NSEQ 4096
#define DD   64
#define BHT  64          // b*h
#define SEG  4           // n-segments for phase-1 parallel reduction
#define SEG_N (NSEQ / SEG)
#define CH   64          // n-chunk per inner iteration
#define NCHUNK (SEG_N / CH)
#define EPSV 1e-10f

typedef __attribute__((ext_vector_type(16))) __bf16       v16bf;
typedef __attribute__((ext_vector_type(8)))  float        v8f;
typedef __attribute__((ext_vector_type(8)))  unsigned int v8u;
typedef __attribute__((ext_vector_type(4)))  unsigned int v4u;
typedef __attribute__((ext_vector_type(8)))  int          v8i;
typedef __attribute__((ext_vector_type(4)))  int          v4i;

union ABFrag { v16bf h; v8u u; };

__device__ __forceinline__ unsigned short f2bf(float f) {
  unsigned int u = __float_as_uint(f);
  u += 0x7FFFu + ((u >> 16) & 1u);          // round-to-nearest-even
  return (unsigned short)(u >> 16);
}

__device__ __forceinline__ float elu1(float x) {
  return (x > 0.0f) ? (x + 1.0f) : __expf(x);
}

// den >= EPSV always (clamped), so the fast hardware reciprocal is safe.
__device__ __forceinline__ float fast_rcp(float x) {
  return __builtin_amdgcn_rcpf(x);          // v_rcp_f32
}

// TDM: 1-D copy of `elems` float32 from global to LDS (ISA 8.3/8.4 packing).
// D# group0: [count=1][lds_addr][global_addr 56:0][type=2 in bits 127:126]
// D# group1: data_size=4B, tensor_dim0=tile_dim0=elems, dim0_stride=elems.
__device__ __forceinline__ void tdm_load_f32(unsigned lds_off,
                                             unsigned long long gaddr,
                                             unsigned elems) {
  v4u g0;
  g0.x = 1u;                                       // count=1, user mode
  g0.y = lds_off;                                  // lds_addr (bytes)
  g0.z = (unsigned)(gaddr & 0xFFFFFFFFull);        // global_addr[31:0]
  g0.w = (unsigned)((gaddr >> 32) & 0x01FFFFFFull) // global_addr[56:32]
         | (2u << 30);                             // type=2 ("image")
  v8i g1;
  g1[0] = (int)(2u << 16);                         // data_size=2 -> 4 bytes
  g1[1] = (int)((elems & 0xFFFFu) << 16);          // tensor_dim0[15:0] @bits63:48
  g1[2] = (int)(((elems >> 16) & 0xFFFFu)          // tensor_dim0[31:16]
                | (1u << 16));                     // tensor_dim1 = 1
  g1[3] = (int)((elems & 0xFFFFu) << 16);          // tile_dim0 @bits127:112
  g1[4] = 0;                                       // tile_dim1/2 = 0 (unused)
  g1[5] = (int)elems;                              // tensor_dim0_stride[31:0]
  g1[6] = 0;
  g1[7] = 0;
  v4i gz4 = {0, 0, 0, 0};
  v8i gz8 = {0, 0, 0, 0, 0, 0, 0, 0};
  __builtin_amdgcn_tensor_load_to_lds(g0, g1, gz4, gz4, gz8, 0);
}

// A fragment (16x32 bf16, row = lane&15). Element j: vgpr p=j>>1,
// k = (p>=4?16:0) + half*8 + 2*(p&3) + (j&1)  -> contiguous bf16 pairs.
__device__ __forceinline__ v16bf load_frag_A(const unsigned short* row, int kb, int half) {
  ABFrag t;
#pragma unroll
  for (int p = 0; p < 8; ++p) {
    int k0 = kb * 32 + ((p & 4) ? 16 : 0) + half * 8 + ((p & 3) << 1);
    t.u[p] = *(const unsigned int*)(row + k0);
  }
  return t.h;
}

// B fragment (32x16 bf16, col = lane&15). Element j: vgpr p=j>>1,
// k = (p>=4?16:0) + half*8 + 2*(p&3) + (j&1)  (same order as A).
__device__ __forceinline__ v16bf load_frag_B(const unsigned short* col, int kb, int half) {
  ABFrag t;
#pragma unroll
  for (int p = 0; p < 8; ++p) {
    int k0 = kb * 32 + ((p & 4) ? 16 : 0) + half * 8 + ((p & 3) << 1);
    t.u[p] = *(const unsigned int*)(col + k0);
  }
  return t.h;
}

__device__ __forceinline__ v8f wmma_bf16(v16bf a, v16bf b, v8f c) {
  return __builtin_amdgcn_wmma_f32_16x16x32_bf16(false, a, false, b, (short)0, c,
                                                 false, false);
}

// ---------------------------------------------------------------------------
// Phase 1: per (segment, bh): accumulate partial new_kv = sum_n kf^T (V - delta)
// and partial new_norm = sum_n kf.
// ---------------------------------------------------------------------------
__global__ __launch_bounds__(512) void fw_phase1(
    const float* __restrict__ keys, const float* __restrict__ values,
    const float* __restrict__ past_kv, const float* __restrict__ past_norm,
    float* __restrict__ pkv_part, float* __restrict__ pn_part) {
  const int seg = blockIdx.x;
  const int bh  = blockIdx.y;
  const float* Kp  = keys      + (size_t)bh * NSEQ * DD;
  const float* Vp  = values    + (size_t)bh * NSEQ * DD;
  const float* PKV = past_kv   + (size_t)bh * DD * DD;
  const float* PN  = past_norm + (size_t)bh * DD;

  __shared__ unsigned short s_kfA[CH * DD];   // k_feat  [n][dk]   (A of delta GEMM)
  __shared__ unsigned short s_kfT[DD * CH];   // k_feat^T[dk][n]   (A of main GEMM)
  __shared__ unsigned short s_vcT[DD * CH];   // v_corr^T[dv][n]   (B of main GEMM)
  __shared__ unsigned short s_pkvT[DD * DD];  // past_kv^T[dv][dk] (B of delta GEMM)
  __shared__ float s_vals[CH * DD];           // raw V tile, filled by the TDM
  __shared__ float s_den[CH];
  __shared__ float s_nacc[DD];

  const int tid  = threadIdx.x;
  const int lane = tid & 31;
  const int wave = tid >> 5;          // 16 waves -> 4x4 tile grid
  const int half = lane >> 4;
  const int l15  = lane & 15;
  const int wi   = wave >> 2;         // n-subtile (delta) / dk-tile (main)
  const int wj   = wave & 3;          // dv-tile

  for (int i = tid; i < DD * DD; i += 512) {
    int dk = i >> 6, dv = i & 63;
    s_pkvT[dv * DD + dk] = f2bf(PKV[i]);
  }
  if (tid < DD) s_nacc[tid] = 0.0f;

  const int nl = tid >> 3;            // chunk row 0..63
  const int k8 = (tid & 7) << 3;      // dk base
  float pn8[8];
#pragma unroll
  for (int j = 0; j < 8; ++j) pn8[j] = PN[k8 + j];
  float nacc[8] = {0, 0, 0, 0, 0, 0, 0, 0};

  v8f acc = {};                       // per-wave 16x16 tile of new_kv partial

  // prologue: fetch key chunk 0 into registers (needs elu, so via VGPRs)
  const size_t rowoff = (size_t)nl * DD + k8;
  const float* kr = Kp + (size_t)seg * SEG_N * DD + rowoff;
  float4 ka = *(const float4*)(kr);
  float4 kx = *(const float4*)(kr + 4);

  const unsigned long long vbase =
      (unsigned long long)(uintptr_t)(Vp + (size_t)seg * SEG_N * DD);
  const unsigned s_vals_off = (unsigned)(uintptr_t)(const void*)s_vals;

  __syncthreads();

  for (int c = 0; c < NCHUNK; ++c) {
    // ---- TDM: DMA this chunk's V tile (16KB contiguous) into s_vals ----
    if (wave == 0)
      tdm_load_f32(s_vals_off, vbase + (unsigned long long)c * CH * DD * 4ull,
                   CH * DD);

    if (tid < CH) s_den[tid] = 0.0f;
    __syncthreads();

    // ---- Stage A: featurize current key chunk (f32 -> bf16, dual layouts) ----
    float kin[8] = {ka.x, ka.y, ka.z, ka.w, kx.x, kx.y, kx.z, kx.w};
    float dpart = 0.0f;
#pragma unroll
    for (int j = 0; j < 8; ++j) {
      float f = elu1(kin[j]);
      unsigned short hb = f2bf(f);
      s_kfA[nl * DD + (k8 + j)] = hb;
      s_kfT[(k8 + j) * CH + nl] = hb;
      dpart += f * pn8[j];
      nacc[j] += f;
    }
    atomicAdd(&s_den[nl], dpart);     // ds_add_f32
    if (wave == 0) __builtin_amdgcn_s_wait_tensorcnt(0);
    __syncthreads();                  // s_vals + s_kf* + s_den now visible

    // ---- Prefetch next key chunk while the WMMA stages run (regs only) ----
    if (c + 1 < NCHUNK) {
      const float* krn = kr + (size_t)(c + 1) * CH * DD;
      ka = *(const float4*)(krn);
      kx = *(const float4*)(krn + 4);
    }

    // ---- Stage B: delta = kf @ past_kv (K = dk = 64, two wmma) ----
    v8f dacc = {};
    {
      const unsigned short* arow = s_kfA + (wi * 16 + l15) * DD;
      const unsigned short* bcol = s_pkvT + (wj * 16 + l15) * DD;
#pragma unroll
      for (int t = 0; t < 2; ++t)
        dacc = wmma_bf16(load_frag_A(arow, t, half), load_frag_B(bcol, t, half), dacc);
    }
    // v_corr = V - delta*rcp(den), stored transposed for the main GEMM's B
#pragma unroll
    for (int r = 0; r < 8; ++r) {
      int nrow = wi * 16 + r + half * 8;
      int dv   = wj * 16 + l15;
      float inv = fast_rcp(fmaxf(s_den[nrow], EPSV));
      float vc  = s_vals[nrow * DD + dv] - dacc[r] * inv;
      s_vcT[dv * CH + nrow] = f2bf(vc);
    }
    __syncthreads();

    // ---- Stage C: acc += kf^T @ v_corr (K = n = 64, two wmma) ----
    {
      const unsigned short* arow = s_kfT + (wi * 16 + l15) * CH;
      const unsigned short* bcol = s_vcT + (wj * 16 + l15) * CH;
#pragma unroll
      for (int t = 0; t < 2; ++t)
        acc = wmma_bf16(load_frag_A(arow, t, half), load_frag_B(bcol, t, half), acc);
    }
    __syncthreads();                  // also protects s_vals before next TDM
  }

  // write partial kv tile (C layout: vgpr r -> row r + half*8, lane -> col)
  float* outp = pkv_part + ((size_t)seg * BHT + bh) * DD * DD;
#pragma unroll
  for (int r = 0; r < 8; ++r) {
    int dk = wi * 16 + r + half * 8;
    int dv = wj * 16 + l15;
    outp[dk * DD + dv] = acc[r];
  }
#pragma unroll
  for (int j = 0; j < 8; ++j) atomicAdd(&s_nacc[k8 + j], nacc[j]);
  __syncthreads();
  if (tid < DD) pn_part[((size_t)seg * BHT + bh) * DD + tid] = s_nacc[tid];
}

// ---------------------------------------------------------------------------
// Phase 1b: reduce segment partials (deterministic tree, no global f32 atomics)
// ---------------------------------------------------------------------------
__global__ __launch_bounds__(256) void fw_reduce(
    const float* __restrict__ past_kv, const float* __restrict__ past_norm,
    const float* __restrict__ pkv_part, const float* __restrict__ pn_part,
    float* __restrict__ new_kv, float* __restrict__ new_norm) {
  const int i = blockIdx.x * blockDim.x + threadIdx.x;
  const int KVTOT = BHT * DD * DD;
  if (i < KVTOT) {
    float s = past_kv[i];
#pragma unroll
    for (int g = 0; g < SEG; ++g) s += pkv_part[(size_t)g * KVTOT + i];
    new_kv[i] = s;
  }
  if (i < BHT * DD) {
    float s = past_norm[i];
#pragma unroll
    for (int g = 0; g < SEG; ++g) s += pn_part[g * BHT * DD + i];
    new_norm[i] = s;
  }
}

// ---------------------------------------------------------------------------
// Phase 2: mem_out = (elu(Q)+1) @ new_kv / denom; gated blend into out_final.
// ---------------------------------------------------------------------------
__global__ __launch_bounds__(512) void fw_phase2(
    const float* __restrict__ queries, const float* __restrict__ out_in,
    const float* __restrict__ head_gates,
    const float* __restrict__ new_kv, const float* __restrict__ new_norm,
    float* __restrict__ out_final) {
  const int bh = blockIdx.y;
  const int h  = bh & 15;
  const int n0 = blockIdx.x * CH;
  const float* Qp  = queries  + (size_t)bh * NSEQ * DD;
  const float* NKV = new_kv   + (size_t)bh * DD * DD;
  const float* NN  = new_norm + (size_t)bh * DD;

  __shared__ unsigned short s_qfA[CH * DD];
  __shared__ unsigned short s_nkvT[DD * DD];
  __shared__ float s_den[CH];

  const int tid  = threadIdx.x;
  const int lane = tid & 31;
  const int wave = tid >> 5;
  const int half = lane >> 4;
  const int l15  = lane & 15;
  const int wi   = wave >> 2;
  const int wj   = wave & 3;

  for (int i = tid; i < DD * DD; i += 512) {
    int dk = i >> 6, dv = i & 63;
    s_nkvT[dv * DD + dk] = f2bf(NKV[i]);
  }
  if (tid < CH) s_den[tid] = 0.0f;

  const int nl = tid >> 3;
  const int k8 = (tid & 7) << 3;
  float nn8[8];
#pragma unroll
  for (int j = 0; j < 8; ++j) nn8[j] = NN[k8 + j];
  __syncthreads();

  // featurize queries chunk
  const float* qr = Qp + (size_t)(n0 + nl) * DD + k8;
  float4 qa = *(const float4*)(qr);
  float4 qb = *(const float4*)(qr + 4);
  float qin[8] = {qa.x, qa.y, qa.z, qa.w, qb.x, qb.y, qb.z, qb.w};
  float dpart = 0.0f;
#pragma unroll
  for (int j = 0; j < 8; ++j) {
    float f = elu1(qin[j]);
    s_qfA[nl * DD + (k8 + j)] = f2bf(f);
    dpart += f * nn8[j];
  }
  atomicAdd(&s_den[nl], dpart);
  __syncthreads();

  v8f dacc = {};
  {
    const unsigned short* arow = s_qfA + (wi * 16 + l15) * DD;
    const unsigned short* bcol = s_nkvT + (wj * 16 + l15) * DD;
#pragma unroll
    for (int t = 0; t < 2; ++t)
      dacc = wmma_bf16(load_frag_A(arow, t, half), load_frag_B(bcol, t, half), dacc);
  }

  const float gate = 1.0f / (1.0f + __expf(-head_gates[h]));
#pragma unroll
  for (int r = 0; r < 8; ++r) {
    int nloc = wi * 16 + r + half * 8;
    int dv   = wj * 16 + l15;
    float inv = fast_rcp(fmaxf(s_den[nloc], EPSV));
    float mem = dacc[r] * inv;
    size_t idx = ((size_t)bh * NSEQ + (n0 + nloc)) * DD + dv;
    out_final[idx] = out_in[idx] * gate + mem * (1.0f - gate);
  }
}

// ---------------------------------------------------------------------------
extern "C" void kernel_launch(void* const* d_in, const int* in_sizes, int n_in,
                              void* d_out, int out_size, void* d_ws, size_t ws_size,
                              hipStream_t stream) {
  (void)in_sizes; (void)n_in; (void)out_size; (void)ws_size;
  const float* keys      = (const float*)d_in[0];
  const float* values    = (const float*)d_in[1];
  const float* queries   = (const float*)d_in[2];
  const float* out_in    = (const float*)d_in[3];
  const float* past_kv   = (const float*)d_in[4];
  const float* past_norm = (const float*)d_in[5];
  const float* gates     = (const float*)d_in[6];

  // d_out = [out_final | new_kv | new_norm] flat, in reference return order
  float* out_final = (float*)d_out;
  float* new_kv    = out_final + (size_t)BHT * NSEQ * DD;   // +16777216
  float* new_norm  = new_kv + (size_t)BHT * DD * DD;        // +262144

  float* pkv_part = (float*)d_ws;                            // SEG*64*4096 f32
  float* pn_part  = pkv_part + (size_t)SEG * BHT * DD * DD;  // SEG*64*64 f32

  fw_phase1<<<dim3(SEG, BHT), 512, 0, stream>>>(keys, values, past_kv, past_norm,
                                                pkv_part, pn_part);
  fw_reduce<<<dim3((BHT * DD * DD + 255) / 256), 256, 0, stream>>>(
      past_kv, past_norm, pkv_part, pn_part, new_kv, new_norm);
  fw_phase2<<<dim3(NSEQ / CH, BHT), 512, 0, stream>>>(queries, out_in, gates,
                                                      new_kv, new_norm, out_final);
}
